// DecoderUnit_61314953118253
// MI455X (gfx1250) — compile-verified, matrix operand order checked by reading, not run on hardware
//
#include <hip/hip_runtime.h>
#include <hip/hip_bf16.h>

// Problem constants (match reference)
#define NB 256      // batch
#define NT 256      // encoder seq len
#define NXD 512     // xDim
#define NSD 512     // sDim
#define NAD 512     // attDim
#define NYD 6625    // vocab

typedef __attribute__((ext_vector_type(16))) __bf16 v16bf;
typedef __attribute__((ext_vector_type(8)))  float  v8f;

// ---------------------------------------------------------------------------
// WMMA fragment loaders (wave32, 16x16x32 bf16).
// A 16x32 (MxK), ISA 7.12.2: lanes 0-15 -> row M=lane, elems 0..7 = K 0..7,
// elems 8..15 = K 16..23; lanes 16-31 -> row M=lane-16, K 8..15 and K 24..31.
// B 32x16 (KxN) mirrors with N in place of M; weights are [N,K] row-major, so a
// B column (fixed n, contiguous k) is a contiguous weight row -> same pattern.
// ---------------------------------------------------------------------------
__device__ __forceinline__ v16bf load_frag(const float* __restrict__ base, int ld) {
  const int lane = threadIdx.x & 31;
  const float* q = base + (size_t)(lane & 15) * ld + ((lane >> 4) << 3);
  v16bf f;
#pragma unroll
  for (int i = 0; i < 8; ++i) f[i] = (__bf16)q[i];
#pragma unroll
  for (int i = 0; i < 8; ++i) f[8 + i] = (__bf16)q[16 + i];
  return f;
}

__device__ __forceinline__ v16bf load_frag_rowclamp(const float* __restrict__ W, int ld,
                                                    int row0, int maxrow, int k0) {
  const int lane = threadIdx.x & 31;
  int row = row0 + (lane & 15);
  row = row < maxrow ? row : maxrow;
  const float* q = W + (size_t)row * ld + k0 + ((lane >> 4) << 3);
  v16bf f;
#pragma unroll
  for (int i = 0; i < 8; ++i) f[i] = (__bf16)q[i];
#pragma unroll
  for (int i = 0; i < 8; ++i) f[8 + i] = (__bf16)q[16 + i];
  return f;
}

// ---------------------------------------------------------------------------
// Generic  D[M,N] = A[M,K] * W[N,K]^T + bias[N]   (bf16 WMMA, f32 accumulate)
// One wave computes a 32x64 output tile: 2 m-tiles x 4 n-tiles = 8 independent
// accumulators.  Software-pipelined: next A/B fragments are loaded into fresh
// registers before the WMMAs consuming the current ones issue, killing the
// WMMA->VALU WAR NOPs and overlapping global_load/cvt with the matrix pipe.
// block = 128 threads = 4 waves (4 consecutive 64-wide n-groups).
// M % 32 == 0, K % 32 == 0 required; N ragged (clamped loads, guarded stores).
// ---------------------------------------------------------------------------
__global__ void gemm_wt_bias(const float* __restrict__ A, int lda,
                             const float* __restrict__ W, int ldw,
                             const float* __restrict__ bias,
                             float* __restrict__ D, int ldd,
                             int N, int K) {
  const int wave = threadIdx.x >> 5;
  const int lane = threadIdx.x & 31;
  const int ngroups = (N + 63) >> 6;                       // 64-wide groups
  const int ng = blockIdx.x * (blockDim.x >> 5) + wave;
  if (ng >= ngroups) return;                               // wave-uniform
  const int m0 = blockIdx.y << 5;                          // 32 rows
  const int n0 = ng << 6;                                  // 64 cols

  v8f acc[2][4] = {};
  // prologue: first fragments
  v16bf a0 = load_frag(A + (size_t)m0 * lda, lda);
  v16bf a1 = load_frag(A + (size_t)(m0 + 16) * lda, lda);
  v16bf w  = load_frag_rowclamp(W, ldw, n0, N - 1, 0);

  for (int k0 = 0; k0 < K; k0 += 32) {
    const int kn = (k0 + 32 < K) ? (k0 + 32) : k0;         // next k (dup on last)
    // prefetch next k-step's A fragments into fresh registers
    const v16bf a0n = load_frag(A + (size_t)m0 * lda + kn, lda);
    const v16bf a1n = load_frag(A + (size_t)(m0 + 16) * lda + kn, lda);
#pragma unroll
    for (int j = 0; j < 4; ++j) {
      // load next B fragment (next n-tile, or next k's first tile) before use
      const v16bf wn = (j < 3)
          ? load_frag_rowclamp(W, ldw, n0 + (j + 1) * 16, N - 1, k0)
          : load_frag_rowclamp(W, ldw, n0, N - 1, kn);
      acc[0][j] = __builtin_amdgcn_wmma_f32_16x16x32_bf16(false, a0, false, w,
                                                          (short)0, acc[0][j], false, false);
      acc[1][j] = __builtin_amdgcn_wmma_f32_16x16x32_bf16(false, a1, false, w,
                                                          (short)0, acc[1][j], false, false);
      w = wn;
    }
    a0 = a0n;
    a1 = a1n;
  }

  const int rbase = m0 + ((lane >> 4) << 3);   // lanes 0-15 -> rows m0..+7, 16-31 -> +8
#pragma unroll
  for (int j = 0; j < 4; ++j) {
    const int col = n0 + j * 16 + (lane & 15);
    if (col < N) {
      const float bv = bias[col];
#pragma unroll
      for (int r = 0; r < 8; ++r) {
        D[(size_t)(rbase + r) * ldd + col]      = acc[0][j][r] + bv;
        D[(size_t)(rbase + 16 + r) * ldd + col] = acc[1][j][r] + bv;
      }
    }
  }
}

// ---------------------------------------------------------------------------
// Fused attention scores:
//   scores[b,t] = sum_a tanh( (x[b,t,:]·xe_w[a,:]) + sProj[b,a] + xe_b[a] ) * we_w[a] + we_b
// One wave handles 16 (b,t) rows; the 32 AD tiles are processed in groups of 4
// (4 independent accumulators), software-pipelined like the GEMM.  Never
// materializes xProj (saves 268 MB of HBM traffic).
// block = 256 threads = 8 waves; grid = (B*T/16)/8 = 512 blocks.
// ---------------------------------------------------------------------------
__global__ void attn_scores(const float* __restrict__ x,
                            const float* __restrict__ sProj,
                            const float* __restrict__ xe_w,
                            const float* __restrict__ xe_b,
                            const float* __restrict__ we_w,
                            const float* __restrict__ we_b,
                            float* __restrict__ scores) {
  const int wave = threadIdx.x >> 5;
  const int lane = threadIdx.x & 31;
  const int mtile = blockIdx.x * 8 + wave;
  const int row0 = mtile << 4;            // flat (b,t) index; tile stays in one b
  const int b = row0 >> 8;                // / NT

  float partial[8];
#pragma unroll
  for (int r = 0; r < 8; ++r) partial[r] = 0.f;

  for (int ng = 0; ng < NAD / 64; ++ng) {        // 8 groups of 4 n-tiles
    v8f acc[4] = {};
    v16bf a = load_frag(x + (size_t)row0 * NXD, NXD);
    v16bf w = load_frag(xe_w + (size_t)(ng * 64) * NXD, NXD);
    for (int k0 = 0; k0 < NXD; k0 += 32) {
      const int kn = (k0 + 32 < NXD) ? (k0 + 32) : k0;
      const v16bf an = load_frag(x + (size_t)row0 * NXD + kn, NXD);
#pragma unroll
      for (int j = 0; j < 4; ++j) {
        const v16bf wn = (j < 3)
            ? load_frag(xe_w + (size_t)(ng * 64 + (j + 1) * 16) * NXD + k0, NXD)
            : load_frag(xe_w + (size_t)(ng * 64) * NXD + kn, NXD);
        acc[j] = __builtin_amdgcn_wmma_f32_16x16x32_bf16(false, a, false, w,
                                                         (short)0, acc[j], false, false);
        w = wn;
      }
      a = an;
    }
#pragma unroll
    for (int j = 0; j < 4; ++j) {
      const int col = ng * 64 + j * 16 + (lane & 15);
      const float sp = sProj[(size_t)b * NAD + col] + xe_b[col];
      const float wv = we_w[col];
#pragma unroll
      for (int r = 0; r < 8; ++r)
        partial[r] += tanhf(acc[j][r] + sp) * wv;
    }
  }

  // Reduce over the 16 columns held in each lane-half (xor 1,2,4,8 stays in-half).
#pragma unroll
  for (int r = 0; r < 8; ++r) {
    float v = partial[r];
#pragma unroll
    for (int off = 1; off < 16; off <<= 1) v += __shfl_xor(v, off, 32);
    partial[r] = v;
  }
  const float wb0 = we_b[0];
  if (lane == 0) {
#pragma unroll
    for (int r = 0; r < 8; ++r) scores[row0 + r] = partial[r] + wb0;
  } else if (lane == 16) {
#pragma unroll
    for (int r = 0; r < 8; ++r) scores[row0 + 8 + r] = partial[r] + wb0;
  }
}

// ---------------------------------------------------------------------------
// Softmax over T + context[b,:] = sum_t alpha[b,t] * x[b,t,:]
// One block per batch row.  Second (and last) full read of x; memory bound.
// ---------------------------------------------------------------------------
__global__ void softmax_context(const float* __restrict__ scores,
                                const float* __restrict__ x,
                                float* __restrict__ context) {
  const int b = blockIdx.x;
  const int t = threadIdx.x;
  __shared__ float red[NT];
  __shared__ float alpha[NT];

  const float sc = scores[b * NT + t];
  red[t] = sc;
  __syncthreads();
  for (int off = NT / 2; off > 0; off >>= 1) {
    if (t < off) red[t] = fmaxf(red[t], red[t + off]);
    __syncthreads();
  }
  const float mx = red[0];
  __syncthreads();
  const float e = __expf(sc - mx);
  red[t] = e;
  __syncthreads();
  for (int off = NT / 2; off > 0; off >>= 1) {
    if (t < off) red[t] += red[t + off];
    __syncthreads();
  }
  const float inv = 1.f / red[0];
  alpha[t] = e * inv;
  __syncthreads();

  const float* xb = x + (size_t)b * NT * NXD;
#pragma unroll 2
  for (int c0 = 0; c0 < NXD / NT; ++c0) {
    const int col = t + c0 * NT;
    float acc = 0.f;
    for (int tt = 0; tt < NT; ++tt) {
      if ((tt & 31) == 0 && tt + 32 < NT)
        __builtin_prefetch(&xb[(size_t)(tt + 32) * NXD + col], 0, 1);
      acc += alpha[tt] * xb[(size_t)tt * NXD + col];
    }
    context[(size_t)b * NXD + col] = acc;
  }
}

// cc = [ emb[yPrev[b]] (AD) | context[b] (XD) ]
__global__ void build_cc(const int* __restrict__ yPrev,
                         const float* __restrict__ emb,
                         const float* __restrict__ context,
                         float* __restrict__ cc) {
  const int i = blockIdx.x * blockDim.x + threadIdx.x;     // over B*(AD+XD)
  const int b = i >> 10;
  const int c = i & 1023;
  float v;
  if (c < NAD) v = emb[(size_t)yPrev[b] * NAD + c];
  else         v = context[(size_t)b * NXD + (c - NAD)];
  cc[i] = v;
}

// GRU gates + hidden update; h written directly into its d_out slot.
__global__ void gru_gate(const float* __restrict__ gi,
                         const float* __restrict__ gh,
                         const float* __restrict__ s,
                         float* __restrict__ h) {
  const int i = blockIdx.x * blockDim.x + threadIdx.x;     // over B*SD
  const int b = i >> 9;
  const int c = i & 511;
  const size_t o = (size_t)b * 3 * NSD;
  const float ir = gi[o + c],            hr = gh[o + c];
  const float iz = gi[o + NSD + c],      hz = gh[o + NSD + c];
  const float in_ = gi[o + 2 * NSD + c], hn = gh[o + 2 * NSD + c];
  const float r = 1.f / (1.f + __expf(-(ir + hr)));
  const float z = 1.f / (1.f + __expf(-(iz + hz)));
  const float n = tanhf(in_ + r * hn);
  const float sv = s[i];
  h[i] = (1.f - z) * n + z * sv;
}

// ---------------------------------------------------------------------------
extern "C" void kernel_launch(void* const* d_in, const int* in_sizes, int n_in,
                              void* d_out, int out_size, void* d_ws, size_t ws_size,
                              hipStream_t stream) {
  const float* x        = (const float*)d_in[0];   // [B,T,XD]
  const float* sPrev    = (const float*)d_in[1];   // [1,B,SD] == [B,SD]
  const int*   yPrev    = (const int*)  d_in[2];   // [B]
  const float* xe_w     = (const float*)d_in[3];   // [AD,XD]
  const float* xe_b     = (const float*)d_in[4];
  const float* se_w     = (const float*)d_in[5];   // [AD,SD]
  const float* se_b     = (const float*)d_in[6];
  const float* we_w     = (const float*)d_in[7];   // [1,AD]
  const float* we_b     = (const float*)d_in[8];   // [1]
  const float* emb      = (const float*)d_in[9];   // [YD+1,AD]
  const float* gru_w_ih = (const float*)d_in[10];  // [3*SD, XD+AD]
  const float* gru_w_hh = (const float*)d_in[11];  // [3*SD, SD]
  const float* gru_b_ih = (const float*)d_in[12];
  const float* gru_b_hh = (const float*)d_in[13];
  const float* fc_w     = (const float*)d_in[14];  // [YD,SD]
  const float* fc_b     = (const float*)d_in[15];

  float* logits = (float*)d_out;                       // [B,YD]
  float* hOut   = (float*)d_out + (size_t)NB * NYD;    // [B,SD]

  // workspace layout (floats)
  float* ws      = (float*)d_ws;
  float* sProj   = ws;                         // B*AD      = 131072
  float* scores  = sProj   + NB * NAD;         // B*T       =  65536
  float* context = scores  + NB * NT;          // B*XD      = 131072
  float* cc      = context + NB * NXD;         // B*(AD+XD) = 262144
  float* gi      = cc      + NB * (NAD + NXD); // B*3*SD    = 393216
  float* gh      = gi      + NB * 3 * NSD;     // B*3*SD    = 393216

  const dim3 blkGemm(128);         // 4 waves -> 4 n-groups (64 cols each) per block
  const int mblocks = NB / 32;     // 32 rows per wave tile

  // 1) sProj = s @ se_w.T + se_b                    [256 x 512 x 512]
  gemm_wt_bias<<<dim3((NAD / 64 + 3) / 4, mblocks), blkGemm, 0, stream>>>(
      sPrev, NSD, se_w, NSD, se_b, sProj, NAD, NAD, NSD);

  // 2) fused xProj + tanh + score dot               [65536 x 512 x 512]
  attn_scores<<<dim3((NB * NT / 16) / 8), dim3(256), 0, stream>>>(
      x, sProj, xe_w, xe_b, we_w, we_b, scores);

  // 3) softmax over T + context
  softmax_context<<<dim3(NB), dim3(NT), 0, stream>>>(scores, x, context);

  // 4) cc = [emb[yPrev], context]
  build_cc<<<dim3(NB * (NAD + NXD) / 256), dim3(256), 0, stream>>>(
      yPrev, emb, context, cc);

  // 5) gi = cc @ gru_w_ih.T + b_ih                  [256 x 1536 x 1024]
  gemm_wt_bias<<<dim3((3 * NSD / 64 + 3) / 4, mblocks), blkGemm, 0, stream>>>(
      cc, NAD + NXD, gru_w_ih, NAD + NXD, gru_b_ih, gi, 3 * NSD, 3 * NSD, NAD + NXD);

  // 6) gh = s @ gru_w_hh.T + b_hh                   [256 x 1536 x 512]
  gemm_wt_bias<<<dim3((3 * NSD / 64 + 3) / 4, mblocks), blkGemm, 0, stream>>>(
      sPrev, NSD, gru_w_hh, NSD, gru_b_hh, gh, 3 * NSD, 3 * NSD, NSD);

  // 7) gates -> h
  gru_gate<<<dim3(NB * NSD / 256), dim3(256), 0, stream>>>(gi, gh, sPrev, hOut);

  // 8) logits = h @ fc_w.T + fc_b                   [256 x 6625 x 512]
  gemm_wt_bias<<<dim3(((NYD + 63) / 64 + 3) / 4, mblocks), blkGemm, 0, stream>>>(
      hOut, NSD, fc_w, NSD, fc_b, logits, NYD, NYD, NSD);
}
